// GNN_68186900791563
// MI455X (gfx1250) — matrix operand order repaired and checked by reading, not compiled, not run on hardware
//
#include <hip/hip_runtime.h>
#include <math.h>

// ---------- types ----------
typedef __attribute__((ext_vector_type(16))) __bf16         v16bf;
typedef __attribute__((ext_vector_type(16))) unsigned short v16us;
typedef __attribute__((ext_vector_type(8)))  float          v8f;
typedef __attribute__((ext_vector_type(4)))  float          f4;

// fp32 -> bf16 (round to nearest even), as raw u16
__device__ __forceinline__ unsigned short f2bfu(float f) {
    unsigned u = __float_as_uint(f);
    u += 0x7FFFu + ((u >> 16) & 1u);
    return (unsigned short)(u >> 16);
}

__device__ __forceinline__ float gelu_exact(float x) {
    return 0.5f * x * (1.0f + erff(x * 0.70710678118654752f));
}

__device__ __forceinline__ void atomicMaxF(float* addr, float val) {
    if (val >= 0.0f) atomicMax((int*)addr, __float_as_int(val));
    else             atomicMin((unsigned int*)addr, __float_as_uint(val));
}

// =====================================================================
// Weight pre-pass: convert fp32 W[T][K][128] to bf16, pre-swizzled into the
// WMMA B-fragment layout so the GEMM loads one aligned 32B vector per lane:
//   Wf[((t*(K/32)+ks)*8 + colTile)*32 + lane][e]  with
//   K-index = ks*32 + e + 16*(lane>>4),  N = colTile*16 + (lane&15)
// =====================================================================
__global__ __launch_bounds__(256) void conv_w_frag(
    const float* __restrict__ W, unsigned short* __restrict__ Wf, int T, int K)
{
    const int P = K * 128;                 // elements per type
    long long idx = (long long)blockIdx.x * 256 + threadIdx.x;
    if (idx >= (long long)T * P) return;
    int t    = (int)(idx / P);
    int idx2 = (int)(idx % P);
    int e    = idx2 & 15;
    int lane = (idx2 >> 4) & 31;
    int ct   = (idx2 >> 9) & 7;
    int ks   = idx2 >> 12;
    int Kk   = ks * 32 + e + 16 * (lane >> 4);
    int N    = ct * 16 + (lane & 15);
    Wf[idx] = f2bfu(W[((size_t)t * K + Kk) * 128 + N]);
}

// =====================================================================
// Per-type GEMM: Y[row,:128] = act( X[row,:DIN] @ W[type(row)] + bias[type] )
// Block = 256 thr (8 waves). Block tile: 32 rows x 128 cols; each wave owns
// two 16x16 WMMA tiles sharing every B fragment (bf16 in, f32 accum).
// mode: 0=none, 1=exact GELU,
//       2=gated skip: Y=(acc+bias)*sig(skip[t]) + Hprev*(1-sig(skip[t]))
// =====================================================================
template <int DIN, int NT>
__global__ __launch_bounds__(256) void pt_gemm(
    const float* __restrict__ X, const int* __restrict__ types,
    const unsigned short* __restrict__ Wf,   // frag layout [NT][DIN/32][8][32][16]
    const float* __restrict__ bias,
    float* __restrict__ Y, int n, int mode,
    const float* __restrict__ skip, const float* __restrict__ Hprev)
{
    __shared__ __align__(16) unsigned short As[32 * DIN];
    __shared__ int tys[32];

    const int row0 = blockIdx.x * 32;

    // stage A tile (32 x DIN) as bf16 into LDS, vectorized float4 loads
    for (int i = threadIdx.x * 4; i < 32 * DIN; i += 1024) {
        int rr = i / DIN, cc = i % DIN;
        int row = row0 + rr;
        f4 v = {0.f, 0.f, 0.f, 0.f};
        if (row < n) v = *(const f4*)(X + (size_t)row * DIN + cc);
        unsigned int lo = (unsigned)f2bfu(v[0]) | ((unsigned)f2bfu(v[1]) << 16);
        unsigned int hi = (unsigned)f2bfu(v[2]) | ((unsigned)f2bfu(v[3]) << 16);
        *(unsigned int*)(&As[i])     = lo;
        *(unsigned int*)(&As[i + 2]) = hi;
    }
    if (threadIdx.x < 32) {
        int row = row0 + threadIdx.x;
        tys[threadIdx.x] = (types && row < n) ? types[row] : 0;
    }
    __syncthreads();

    const int wv   = threadIdx.x >> 5;
    const int lane = threadIdx.x & 31;
    const int half = lane >> 4;
    const int nn   = lane & 15;
    const int m    = lane & 15;

    v8f acc[NT][2];
    const v8f vzero = {0.f, 0.f, 0.f, 0.f, 0.f, 0.f, 0.f, 0.f};
#pragma unroll
    for (int t = 0; t < NT; ++t) { acc[t][0] = vzero; acc[t][1] = vzero; }

#pragma unroll
    for (int ks = 0; ks < DIN / 32; ++ks) {
        // two A fragments (rows m and m+16), gathered from LDS (b128 pairs)
        v16us au0, au1;
#pragma unroll
        for (int e = 0; e < 16; ++e) {
            int K = (half << 4) + e;
            au0[e] = As[m * DIN + ks * 32 + K];
            au1[e] = As[(m + 16) * DIN + ks * 32 + K];
        }
        v16bf a0 = __builtin_bit_cast(v16bf, au0);
        v16bf a1 = __builtin_bit_cast(v16bf, au1);

#pragma unroll
        for (int t = 0; t < NT; ++t) {
            // one aligned 32B vector load per lane: the whole B fragment
            v16us bu = *((const v16us*)Wf +
                         (((size_t)t * (DIN / 32) + ks) * 8 + wv) * 32 + lane);
            v16bf b = __builtin_bit_cast(v16bf, bu);
            acc[t][0] = __builtin_amdgcn_wmma_f32_16x16x32_bf16(
                false, a0, false, b, (short)0, acc[t][0], false, false);
            acc[t][1] = __builtin_amdgcn_wmma_f32_16x16x32_bf16(
                false, a1, false, b, (short)0, acc[t][1], false, false);
        }
    }

    const int col = wv * 16 + nn;
#pragma unroll
    for (int mt = 0; mt < 2; ++mt) {
#pragma unroll
        for (int g = 0; g < 8; ++g) {
            int mm  = mt * 16 + g + 8 * half;  // C/D layout: VGPR g <-> row
            int row = row0 + mm;
            if (row >= n) continue;
            int t = (NT > 1) ? (tys[mm] & 3) : 0;
            float v;
            switch (t) {
                case 0:  v = acc[0][mt][g]; break;
                case 1:  v = acc[(NT > 1) ? 1 : 0][mt][g]; break;
                case 2:  v = acc[(NT > 2) ? 2 : 0][mt][g]; break;
                default: v = acc[(NT > 3) ? 3 : 0][mt][g]; break;
            }
            if (bias) v += bias[t * 128 + col];
            if (mode == 1) {
                v = gelu_exact(v);
            } else if (mode == 2) {
                float s = 1.0f / (1.0f + expf(-skip[t]));
                v = v * s + Hprev[(size_t)row * 128 + col] * (1.0f - s);
            }
            Y[(size_t)row * 128 + col] = v;
        }
    }
}

// =====================================================================
// Edge pass A: attention scores + segment max.
// One wave per edge: k_e = k_node[src] + rte_k[st][time] staged in LDS;
// att[h] = (q[dst] . (k_e @ rel_att[r,h])) * pri[r,h] / sqrt(32)
// =====================================================================
__global__ __launch_bounds__(256) void edge_att_kernel(
    const int* __restrict__ src, const int* __restrict__ dst,
    const int* __restrict__ etype, const int* __restrict__ etime,
    const int* __restrict__ ntype,
    const float* __restrict__ k_node, const float* __restrict__ q_node,
    const float* __restrict__ rte_k,           // [4,240,128]
    const float* __restrict__ RA,              // [5,4,32,32] (layer slice)
    const float* __restrict__ pri,             // [5,4]
    float* __restrict__ att, float* __restrict__ amax, int nE)
{
    __shared__ float ksh[8][128];
    const int w = threadIdx.x >> 5, lane = threadIdx.x & 31;
    const int e0 = blockIdx.x * 8 + w;
    const bool valid = e0 < nE;
    const int e = valid ? e0 : 0;

    const int s = src[e], d = dst[e], r = etype[e], tm = etime[e];
    const int st = ntype[s];
    const float* kn = k_node + (size_t)s * 128;
    const float* rk = rte_k + ((size_t)st * 240 + tm) * 128;
    for (int i = lane; i < 128; i += 32) ksh[w][i] = kn[i] + rk[i];
    __syncthreads();

    const int h = lane >> 3, jb = (lane & 7) * 4;
    const float* A = RA + (size_t)(r * 4 + h) * 1024;
    float attv = 0.0f;
#pragma unroll
    for (int jj = 0; jj < 4; ++jj) {
        int j = jb + jj;
        float kr = 0.0f;
#pragma unroll
        for (int i = 0; i < 32; ++i)
            kr = fmaf(ksh[w][h * 32 + i], A[i * 32 + j], kr);
        attv = fmaf(q_node[(size_t)d * 128 + h * 32 + j], kr, attv);
    }
    attv += __shfl_xor(attv, 1, 32);
    attv += __shfl_xor(attv, 2, 32);
    attv += __shfl_xor(attv, 4, 32);
    if (valid && (lane & 7) == 0) {
        float a = attv * pri[r * 4 + h] * 0.17677669529663687f; // 1/sqrt(32)
        att[(size_t)e * 4 + h] = a;
        atomicMaxF(&amax[(size_t)d * 4 + h], a);
    }
}

// =====================================================================
// Edge pass B: ex = exp(att - amax[dst]); denom += ex;
// aggr[dst] += ex * (v_e @ rel_msg[r,h])  (unnormalized; normalized later)
// =====================================================================
__global__ __launch_bounds__(256) void edge_msg_kernel(
    const int* __restrict__ src, const int* __restrict__ dst,
    const int* __restrict__ etype, const int* __restrict__ etime,
    const int* __restrict__ ntype,
    const float* __restrict__ v_node, const float* __restrict__ rte_v,
    const float* __restrict__ RM,              // [5,4,32,32]
    const float* __restrict__ att, const float* __restrict__ amax,
    float* __restrict__ denom, float* __restrict__ aggr, int nE)
{
    __shared__ float vsh[8][128];
    const int w = threadIdx.x >> 5, lane = threadIdx.x & 31;
    const int e0 = blockIdx.x * 8 + w;
    const bool valid = e0 < nE;
    const int e = valid ? e0 : 0;

    const int s = src[e], d = dst[e], r = etype[e], tm = etime[e];
    const int st = ntype[s];
    const float* vn = v_node + (size_t)s * 128;
    const float* rv = rte_v + ((size_t)st * 240 + tm) * 128;
    for (int i = lane; i < 128; i += 32) vsh[w][i] = vn[i] + rv[i];
    __syncthreads();

    const int h = lane >> 3, jb = (lane & 7) * 4;
    const float ex = expf(att[(size_t)e * 4 + h] - amax[(size_t)d * 4 + h]);
    if (valid && (lane & 7) == 0) atomicAdd(&denom[(size_t)d * 4 + h], ex);

    const float* M = RM + (size_t)(r * 4 + h) * 1024;
#pragma unroll
    for (int jj = 0; jj < 4; ++jj) {
        int j = jb + jj;
        float mg = 0.0f;
#pragma unroll
        for (int i = 0; i < 32; ++i)
            mg = fmaf(vsh[w][h * 32 + i], M[i * 32 + j], mg);
        if (valid) atomicAdd(&aggr[(size_t)d * 128 + h * 32 + j], ex * mg);
    }
}

__global__ __launch_bounds__(256) void init_reduce_bufs(
    float* __restrict__ amax, float* __restrict__ denom,
    float* __restrict__ aggr, int n4, int nagg)
{
    int i = blockIdx.x * 256 + threadIdx.x;
    if (i < n4) { amax[i] = -3.402823466e38f; denom[i] = 0.0f; }
    if (i < nagg) aggr[i] = 0.0f;
}

// aggr <- gelu(aggr / (denom + 1e-16)) in place
__global__ __launch_bounds__(256) void finish_gelu_kernel(
    float* __restrict__ aggr, const float* __restrict__ denom, int nNodes)
{
    int i = blockIdx.x * 256 + threadIdx.x;
    if (i >= nNodes * 128) return;
    int node = i >> 7;
    int head = (i >> 5) & 3;
    float x = aggr[i] / (denom[node * 4 + head] + 1e-16f);
    aggr[i] = gelu_exact(x);
}

// =====================================================================
extern "C" void kernel_launch(void* const* d_in, const int* in_sizes, int n_in,
                              void* d_out, int out_size, void* d_ws, size_t ws_size,
                              hipStream_t stream)
{
    (void)in_sizes; (void)n_in; (void)out_size; (void)ws_size;
    constexpr int NN = 50000, NE = 400000, DI = 256, NH = 128;

    const float* node_feature = (const float*)d_in[0];
    const float* adapt_W      = (const float*)d_in[1];
    const float* adapt_b      = (const float*)d_in[2];
    const float* Wk           = (const float*)d_in[3];
    const float* bk           = (const float*)d_in[4];
    const float* Wq           = (const float*)d_in[5];
    const float* bq           = (const float*)d_in[6];
    const float* Wv           = (const float*)d_in[7];
    const float* bv           = (const float*)d_in[8];
    const float* Wa           = (const float*)d_in[9];
    const float* ba           = (const float*)d_in[10];
    const float* rel_pri      = (const float*)d_in[11];
    const float* rel_att      = (const float*)d_in[12];
    const float* rel_msg      = (const float*)d_in[13];
    const float* skip         = (const float*)d_in[14];
    const float* rte_emb      = (const float*)d_in[15];
    const float* rte_W        = (const float*)d_in[16];
    const float* rte_b        = (const float*)d_in[17];
    const int*   node_type    = (const int*)d_in[18];
    const int*   edge_index   = (const int*)d_in[19];
    const int*   edge_type    = (const int*)d_in[20];
    const int*   edge_time    = (const int*)d_in[21];
    const int* srcp = edge_index;
    const int* dstp = edge_index + NE;

    float* ws = (float*)d_ws;
    size_t o = 0;
    auto alloc = [&](size_t nf) {                 // 64B-aligned sub-allocs
        float* p = ws + o; o += (nf + 15) & ~(size_t)15; return p;
    };
    float* hA       = alloc((size_t)NN * NH);
    float* hB       = alloc((size_t)NN * NH);
    float* k_node   = alloc((size_t)NN * NH);
    float* q_node   = alloc((size_t)NN * NH);
    float* v_node   = alloc((size_t)NN * NH);
    float* rte_base = alloc((size_t)240 * NH);
    float* rte_k    = alloc((size_t)4 * 240 * NH);
    float* rte_v    = alloc((size_t)4 * 240 * NH);
    float* attb     = alloc((size_t)NE * 4);
    float* amax     = alloc((size_t)NN * 4);
    float* denom    = alloc((size_t)NN * 4);
    float* aggr     = alloc((size_t)NN * NH);
    // bf16 swizzled weight fragments (ushort counts / 2 = float counts)
    unsigned short* fadapt = (unsigned short*)alloc((size_t)4 * DI * NH / 2);
    unsigned short* fWk    = (unsigned short*)alloc((size_t)8 * NH * NH / 2);
    unsigned short* fWq    = (unsigned short*)alloc((size_t)8 * NH * NH / 2);
    unsigned short* fWv    = (unsigned short*)alloc((size_t)8 * NH * NH / 2);
    unsigned short* fWa    = (unsigned short*)alloc((size_t)8 * NH * NH / 2);
    unsigned short* frteW  = (unsigned short*)alloc((size_t)2 * NH * NH / 2);

    const dim3 blk(256);
    const int gN = (NN + 31) / 32;   // 1563
    const int gR = (240 + 31) / 32;  // 8
    const int gE = NE / 8;           // 50000
    const int gElem = (NN * NH + 255) / 256;
    const size_t fstride = (size_t)NH * NH;   // ushorts per 128x128 type slice

    // ---- one-shot weight conversion + swizzle (bf16 B-fragment layout) ----
    conv_w_frag<<<(4 * DI * NH + 255) / 256, blk, 0, stream>>>(adapt_W, fadapt, 4, DI);
    conv_w_frag<<<(8 * NH * NH + 255) / 256, blk, 0, stream>>>(Wk, fWk, 8, NH);
    conv_w_frag<<<(8 * NH * NH + 255) / 256, blk, 0, stream>>>(Wq, fWq, 8, NH);
    conv_w_frag<<<(8 * NH * NH + 255) / 256, blk, 0, stream>>>(Wv, fWv, 8, NH);
    conv_w_frag<<<(8 * NH * NH + 255) / 256, blk, 0, stream>>>(Wa, fWa, 8, NH);
    conv_w_frag<<<(2 * NH * NH + 255) / 256, blk, 0, stream>>>(rte_W, frteW, 2, NH);

    // input adaptation: h = gelu(per-type Linear 256->128)
    pt_gemm<DI, 4><<<gN, blk, 0, stream>>>(node_feature, node_type, fadapt,
                                           adapt_b, hA, NN, 1, nullptr, nullptr);

    for (int l = 0; l < 2; ++l) {
        const float* h    = (l == 0) ? hA : hB;
        float*       hout = (l == 0) ? hB : (float*)d_out;
        const float* bk_l = bk + (size_t)l * 4 * NH;
        const float* bq_l = bq + (size_t)l * 4 * NH;
        const float* bv_l = bv + (size_t)l * 4 * NH;
        const float* ba_l = ba + (size_t)l * 4 * NH;
        const float* RA_l = rel_att + (size_t)l * 5 * 4 * 32 * 32;
        const float* RM_l = rel_msg + (size_t)l * 5 * 4 * 32 * 32;
        const float* pri_l = rel_pri + (size_t)l * 5 * 4;
        const unsigned short* fWk_l = fWk + (size_t)l * 4 * fstride;
        const unsigned short* fWq_l = fWq + (size_t)l * 4 * fstride;
        const unsigned short* fWv_l = fWv + (size_t)l * 4 * fstride;
        const unsigned short* fWa_l = fWa + (size_t)l * 4 * fstride;

        // RTE base table: rte_base = rte_emb @ rte_W[l] + rte_b[l]  [240,128]
        pt_gemm<NH, 1><<<gR, blk, 0, stream>>>(rte_emb, nullptr,
            frteW + (size_t)l * fstride, rte_b + (size_t)l * NH,
            rte_base, 240, 0, nullptr, nullptr);
        // typed RTE projections: rte_k[t] = rte_base @ Wk[l,t] (no bias)
        for (int t = 0; t < 4; ++t) {
            pt_gemm<NH, 1><<<gR, blk, 0, stream>>>(rte_base, nullptr,
                fWk_l + (size_t)t * fstride, nullptr,
                rte_k + (size_t)t * 240 * NH, 240, 0, nullptr, nullptr);
            pt_gemm<NH, 1><<<gR, blk, 0, stream>>>(rte_base, nullptr,
                fWv_l + (size_t)t * fstride, nullptr,
                rte_v + (size_t)t * 240 * NH, 240, 0, nullptr, nullptr);
        }
        // per-node typed projections
        pt_gemm<NH, 4><<<gN, blk, 0, stream>>>(h, node_type, fWk_l, bk_l,
                                               k_node, NN, 0, nullptr, nullptr);
        pt_gemm<NH, 4><<<gN, blk, 0, stream>>>(h, node_type, fWq_l, bq_l,
                                               q_node, NN, 0, nullptr, nullptr);
        pt_gemm<NH, 4><<<gN, blk, 0, stream>>>(h, node_type, fWv_l, bv_l,
                                               v_node, NN, 0, nullptr, nullptr);

        init_reduce_bufs<<<gElem, blk, 0, stream>>>(amax, denom, aggr,
                                                    NN * 4, NN * NH);
        edge_att_kernel<<<gE, blk, 0, stream>>>(srcp, dstp, edge_type, edge_time,
            node_type, k_node, q_node, rte_k, RA_l, pri_l, attb, amax, NE);
        edge_msg_kernel<<<gE, blk, 0, stream>>>(srcp, dstp, edge_type, edge_time,
            node_type, v_node, rte_v, RM_l, attb, amax, denom, aggr, NE);
        finish_gelu_kernel<<<gElem, blk, 0, stream>>>(aggr, denom, NN);

        // output transform + gated skip into hout
        pt_gemm<NH, 4><<<gN, blk, 0, stream>>>(aggr, node_type, fWa_l, ba_l,
                                               hout, NN, 2, skip + (size_t)l * 4, h);
    }
}
